// Model_19052474925202
// MI455X (gfx1250) — matrix operand order repaired and checked by reading, not compiled, not run on hardware
//
#include <hip/hip_runtime.h>
#include <hip/hip_bf16.h>
#include <math.h>

// ---------------- types ----------------
typedef _Float16 v16h __attribute__((ext_vector_type(16)));
typedef _Float16 v8h  __attribute__((ext_vector_type(8)));
typedef float    v8f  __attribute__((ext_vector_type(8)));

// global-address-space element types (force global_load_*, not flat_load_*)
typedef __attribute__((address_space(1))) const _Float16 GF16;
typedef __attribute__((address_space(1))) const float    GFLT;
typedef __attribute__((address_space(1))) const v8h      GV8H;

union V16U { v16h v; v8h h[2]; };

// ---------------- model dims ----------------
#define B_TOT   512
#define L_T     128
#define H_D     128
#define N_EDGES 50000
#define ADJ_K   9

// packed weight offsets (in halves) inside workspace
#define OFF_W0   0          // [512][320]
#define OFF_W1   163840     // [512][256]
#define OFF_W2   294912     // [512][256]
#define OFF_WAX  425984     // [32][128]
#define OFF_WAT  430080     // [32][64]
#define OFF_WMX  432128     // [32][192]
#define OFF_WMT  438272     // [32][192]
#define HALVES_TOTAL 444416
#define FB_BYTE_OFF  (HALVES_TOTAL * 2)   // 888832, 16B aligned
#define FLOATS_TOTAL 1664

// LDS layout (bytes)
#define SM_A0    0          // [16][320] f16
#define SM_A1    10240      // [16][256] f16
#define SM_A2    18432      // [16][256] f16
#define SM_GATES 26624      // [16][512] f32
#define SM_CBUF  59392      // 3*[16][128] f32
#define SM_XTS   83968      // 16 int
#define SM_DTS   84032      // 16 int
#define SM_ADJ   84096      // 144 int (pad 640)
#define SM_MK    84736      // [144][192] f16
#define SM_TQ    140032     // [144][64] f16
#define SM_AFX   158464     // [16][32] f32
#define SM_BUFA  160512     // [144][32] f32
#define SM_BUFB  178944     // [144][32] f32
#define SM_PXS   197376     // [16][9] f32
#define SMEM_BYTES 197952

// ---------------- anti-hoist laundering ----------------
// Weights are L2-resident (packed ~0.9MB shared by all 32 WGs vs 192MB L2), so
// reloading B fragments every timestep is nearly free, while keeping them live
// across the 128-step recurrence spills (seen round 1). Launder the base
// pointer through an opaque SGPR-constrained asm inside the t-loop so LLVM
// cannot hoist fragment loads across timesteps, then re-assert address space 1
// so the loads lower to global_load_b128 (LOADcnt only) instead of
// flat_load_b128 (LOADcnt+DScnt, contends with the LDS A-fragment path).
__device__ __forceinline__ GF16* opaque_g(const _Float16* p) {
    unsigned long long u = (unsigned long long)p;
    asm volatile("" : "+s"(u));
    return (GF16*)u;
}
__device__ __forceinline__ GFLT* opaque_gf(const float* p) {
    unsigned long long u = (unsigned long long)p;
    asm volatile("" : "+s"(u));
    return (GFLT*)u;
}

// ---------------- WMMA fragment helpers ----------------
// A fragment: 16x32 f16 tile from row-major LDS buffer.
// ISA layout: lane(lo,hi): VGPR0-3 = A[lo][hi*8 + 0..7], VGPR4-7 = A[lo][16 + hi*8 + 0..7]
__device__ __forceinline__ v16h load_a_frag(const _Float16* A, int lda, int m0, int k0, int lane) {
    int lo = lane & 15, hi = lane >> 4;
    const _Float16* row = A + (size_t)(m0 + lo) * lda + k0;
    V16U u;
    u.h[0] = *(const v8h*)(row + hi * 8);
    u.h[1] = *(const v8h*)(row + 16 + hi * 8);
    return u.v;
}

// B fragment: 32x16 (KxN) where B = W^T, W packed row-major [N][ldk] f16 in global.
// lane(lo,hi) holds W[n0+lo][k0 + hi*8 + 0..7] then W[n0+lo][k0 + 16 + hi*8 + 0..7] (matches A layout).
__device__ __forceinline__ v16h load_b_frag(GF16* W, int ldk, int n0, int k0, int lane) {
    int lo = lane & 15, hi = lane >> 4;
    GF16* row = W + (size_t)(n0 + lo) * ldk + k0;
    V16U u;
    u.h[0] = *(GV8H*)(row + hi * 8);
    u.h[1] = *(GV8H*)(row + 16 + hi * 8);
    return u.v;
}

template <int KC>
__device__ __forceinline__ void wmma_tiles(const _Float16* A, int lda, int m0,
                                           GF16* W, int ldk, int n0,
                                           int lane, v8f& acc) {
#pragma unroll
    for (int c = 0; c < KC; ++c) {
        v16h a = load_a_frag(A, lda, m0, c * 32, lane);
        v16h b = load_b_frag(W, ldk, n0, c * 32, lane);
        acc = __builtin_amdgcn_wmma_f32_16x16x32_f16(
            /*neg_a=*/false, a, /*neg_b=*/false, b,
            /*c_mod=*/(short)0, acc, /*reuse_a=*/false, /*reuse_b=*/false);
    }
}

// D store: lane(lo,hi), acc[r] = D[m0 + hi*8 + r][n0 + lo]; fused bias (+ optional relu).
__device__ __forceinline__ void store_d(float* Dst, int ldd, int m0, int n0, int lane,
                                        const v8f& acc, GFLT* bias, int relu) {
    int lo = lane & 15, hi = lane >> 4;
    float bv = bias[n0 + lo];
#pragma unroll
    for (int r = 0; r < 8; ++r) {
        float v = acc[r] + bv;
        if (relu) v = v > 0.f ? v : 0.f;
        Dst[(m0 + hi * 8 + r) * ldd + n0 + lo] = v;
    }
}

// LSTM gate nonlinearity + state update; h written as f16 into next-layer A buffers.
__device__ __forceinline__ void lstm_update(const float* gates, float* cb,
                                            _Float16* hdst0, int ld0,
                                            _Float16* hdst1, int ld1, int tid) {
    for (int i = tid; i < 16 * H_D; i += 256) {
        int m = i >> 7, j = i & 127;
        float gi = gates[m * 512 + j];
        float gf = gates[m * 512 + 128 + j];
        float gg = gates[m * 512 + 256 + j];
        float go = gates[m * 512 + 384 + j];
        float c  = cb[i];
        float si = 1.f / (1.f + __expf(-gi));
        float sf = 1.f / (1.f + __expf(-gf));
        float so = 1.f / (1.f + __expf(-go));
        float cn = sf * c + si * tanhf(gg);
        float h  = so * tanhf(cn);
        cb[i] = cn;
        hdst0[m * ld0 + j] = (_Float16)h;
        if (hdst1) hdst1[m * ld1 + j] = (_Float16)h;
    }
}

// ---------------- prep: pack f32 weights -> f16 WMMA-friendly, pre-sum biases ----------------
__global__ void prep_kernel(const float* w_ih0, const float* w_hh0, const float* b_ih0, const float* b_hh0,
                            const float* w_ih1, const float* w_hh1, const float* b_ih1, const float* b_hh1,
                            const float* w_ih2, const float* w_hh2, const float* b_ih2, const float* b_hh2,
                            const float* wax, const float* bax, const float* wat, const float* bat,
                            const float* wmx, const float* bmx, const float* wmt, const float* bmt,
                            char* ws) {
    _Float16* H = (_Float16*)ws;
    float*    F = (float*)(ws + FB_BYTE_OFF);
    int total = HALVES_TOTAL + FLOATS_TOTAL;
    for (int i = blockIdx.x * blockDim.x + threadIdx.x; i < total;
         i += gridDim.x * blockDim.x) {
        if (i < HALVES_TOTAL) {
            float v = 0.f;
            int j = i;
            if (j < OFF_W1) {                       // W0: [512][320] = w_ih0(184,pad192) || w_hh0(128)
                int n = j / 320, k = j % 320;
                if (k < 184)       v = w_ih0[n * 184 + k];
                else if (k >= 192) v = w_hh0[n * 128 + (k - 192)];
            } else if (j < OFF_W2) {                // W1: [512][256]
                j -= OFF_W1; int n = j / 256, k = j % 256;
                v = (k < 128) ? w_ih1[n * 128 + k] : w_hh1[n * 128 + (k - 128)];
            } else if (j < OFF_WAX) {               // W2
                j -= OFF_W2; int n = j / 256, k = j % 256;
                v = (k < 128) ? w_ih2[n * 128 + k] : w_hh2[n * 128 + (k - 128)];
            } else if (j < OFF_WAT) { v = wax[j - OFF_WAX]; }
            else if (j < OFF_WMX)   { v = wat[j - OFF_WAT]; }
            else if (j < OFF_WMT) {                 // Wmx: [32][192] (168 pad)
                j -= OFF_WMX; int n = j / 192, k = j % 192;
                if (k < 168) v = wmx[n * 168 + k];
            } else {
                j -= OFF_WMT; int n = j / 192, k = j % 192;
                if (k < 168) v = wmt[n * 168 + k];
            }
            H[i] = (_Float16)v;
        } else {
            int j = i - HALVES_TOTAL;
            float v;
            if (j < 512)       v = b_ih0[j] + b_hh0[j];
            else if (j < 1024) v = b_ih1[j - 512] + b_hh1[j - 512];
            else if (j < 1536) v = b_ih2[j - 1024] + b_hh2[j - 1024];
            else if (j < 1568) v = bax[j - 1536];
            else if (j < 1600) v = bat[j - 1568];
            else if (j < 1632) v = bmx[j - 1600];
            else               v = bmt[j - 1632];
            F[j] = v;
        }
    }
}

// ---------------- main persistent recurrence kernel ----------------
// grid = 32 blocks (16 batch rows each), block = 256 threads = 8 wave32.
__global__ __launch_bounds__(256) void model_kernel(
    const int* __restrict__ x, const int* __restrict__ dpt,
    const float* __restrict__ z, const int* __restrict__ adj,
    const float* __restrict__ eprop, const float* __restrict__ x_emb,
    const float* __restrict__ t_emb, const float* __restrict__ tcost_emb,
    const char* __restrict__ ws, float* __restrict__ out) {
    extern __shared__ char smem[];
    _Float16* A0  = (_Float16*)(smem + SM_A0);   // [16][320]: z|x_emb|t_emb|eprop|pad | h0
    _Float16* A1  = (_Float16*)(smem + SM_A1);   // [16][256]: h0 | h1
    _Float16* A2  = (_Float16*)(smem + SM_A2);   // [16][256]: h1 | h2
    float* gates  = (float*)(smem + SM_GATES);   // [16][512]
    float* cbuf   = (float*)(smem + SM_CBUF);    // 3*[16][128]
    int* xts      = (int*)(smem + SM_XTS);
    int* dts      = (int*)(smem + SM_DTS);
    int* adjsh    = (int*)(smem + SM_ADJ);       // [144]
    _Float16* MK  = (_Float16*)(smem + SM_MK);   // [144][192]
    _Float16* TQ  = (_Float16*)(smem + SM_TQ);   // [144][64]
    float* afx    = (float*)(smem + SM_AFX);     // [16][32]
    float* BUFA   = (float*)(smem + SM_BUFA);    // [144][32]
    float* BUFB   = (float*)(smem + SM_BUFB);    // [144][32]
    float* pxs    = (float*)(smem + SM_PXS);     // [16][9]

    const _Float16* W0base  = (const _Float16*)ws;
    const float*    Fb      = (const float*)(ws + FB_BYTE_OFF);

    const int tid  = threadIdx.x;
    const int lane = tid & 31;
    const int wave = tid >> 5;
    const int bb   = blockIdx.x * 16;

    // zero recurrent state (h regions + c)
    for (int i = tid; i < 16 * 320; i += 256) A0[i] = (_Float16)0.f;
    for (int i = tid; i < 16 * 256; i += 256) { A1[i] = (_Float16)0.f; A2[i] = (_Float16)0.f; }
    for (int i = tid; i < 3 * 16 * 128; i += 256) cbuf[i] = 0.f;
    __syncthreads();

    for (int t = 0; t < L_T; ++t) {
        // re-materialize (opaque, global-AS) weight/bias pointers each step:
        // loads below are NOT loop-invariant => no hoisting => no spills.
        GF16* W0  = opaque_g(W0base + OFF_W0);
        GF16* W1  = opaque_g(W0base + OFF_W1);
        GF16* W2  = opaque_g(W0base + OFF_W2);
        GF16* Wax = opaque_g(W0base + OFF_WAX);
        GF16* Wat = opaque_g(W0base + OFF_WAT);
        GF16* Wmx = opaque_g(W0base + OFF_WMX);
        GF16* Wmt = opaque_g(W0base + OFF_WMT);
        GFLT* B0g = opaque_gf(Fb);
        GFLT* B1g = opaque_gf(Fb + 512);
        GFLT* B2g = opaque_gf(Fb + 1024);
        GFLT* Bax = opaque_gf(Fb + 1536);
        GFLT* Bat = opaque_gf(Fb + 1568);
        GFLT* Bmx = opaque_gf(Fb + 1600);
        GFLT* Bmt = opaque_gf(Fb + 1632);

        if (tid < 16) {
            xts[tid] = x[(bb + tid) * L_T + t];
            dts[tid] = dpt[(bb + tid) * L_T + t];
        }
        __syncthreads();
        // gather/convert layer-0 input: [z(16) | x_emb(128) | t_emb(32) | eprop(8) | pad(8)]
        for (int i = tid; i < 16 * 192; i += 256) {
            int r = i / 192, j = i - r * 192;
            float v;
            if (j < 16)       v = z[((size_t)t * B_TOT + bb + r) * 16 + j];
            else if (j < 144) v = x_emb[(size_t)xts[r] * 128 + (j - 16)];
            else if (j < 176) v = t_emb[dts[r] * 32 + (j - 144)];
            else if (j < 184) v = eprop[(size_t)xts[r] * 8 + (j - 176)];
            else              v = 0.f;
            A0[r * 320 + j] = (_Float16)v;
        }
        __syncthreads();

        // ---- LSTM layer 0: [16,320] x [320,512] ----
        for (int nt = 0; nt < 4; ++nt) {
            int n0 = (wave * 4 + nt) * 16;
            v8f acc = {};
            wmma_tiles<10>(A0, 320, 0, W0, 320, n0, lane, acc);
            store_d(gates, 512, 0, n0, lane, acc, B0g, 0);
        }
        __syncthreads();
        lstm_update(gates, cbuf, A0 + 192, 320, A1, 256, tid);
        __syncthreads();

        // ---- LSTM layer 1: [16,256] x [256,512] ----
        for (int nt = 0; nt < 4; ++nt) {
            int n0 = (wave * 4 + nt) * 16;
            v8f acc = {};
            wmma_tiles<8>(A1, 256, 0, W1, 256, n0, lane, acc);
            store_d(gates, 512, 0, n0, lane, acc, B1g, 0);
        }
        __syncthreads();
        lstm_update(gates, cbuf + 2048, A1 + 128, 256, A2, 256, tid);
        __syncthreads();

        // ---- LSTM layer 2 ----
        for (int nt = 0; nt < 4; ++nt) {
            int n0 = (wave * 4 + nt) * 16;
            v8f acc = {};
            wmma_tiles<8>(A2, 256, 0, W2, 256, n0, lane, acc);
            store_d(gates, 512, 0, n0, lane, acc, B2g, 0);
        }
        __syncthreads();
        lstm_update(gates, cbuf + 4096, A2 + 128, 256, (_Float16*)nullptr, 0, tid);
        __syncthreads();

        // ---- heads: gather adjacency, build MK [144,192] and TQ [144,64] ----
        for (int i = tid; i < 144; i += 256)
            adjsh[i] = adj[(size_t)xts[i / 9] * ADJ_K + (i % 9)];
        __syncthreads();
        for (int i = tid; i < 144 * 192; i += 256) {
            int rho = i / 192, j = i - rho * 192, r = rho / 9;
            int e = adjsh[rho];
            _Float16 v;
            if (j < 128)      v = (_Float16)x_emb[(size_t)e * 128 + j];
            else if (j < 160) v = A0[r * 320 + 144 + (j - 128)];       // tt (already f16)
            else if (j < 168) v = (_Float16)eprop[(size_t)e * 8 + (j - 160)];
            else              v = (_Float16)0.f;
            MK[rho * 192 + j] = v;
        }
        for (int i = tid; i < 144 * 64; i += 256) {
            int rho = i >> 6, j = i & 63, r = rho / 9;
            int e = adjsh[rho];
            TQ[rho * 64 + j] = (j < 32) ? (_Float16)tcost_emb[(size_t)e * 32 + j]
                                        : A0[r * 320 + 144 + (j - 32)];
        }
        __syncthreads();

        // jobs 0..1: afterx = relu(h2 @ Wax^T + bax)   [16,32]
        // jobs 2..19: BUFA = relu(TQ @ Wat^T + bat)    [144,32]
        // jobs 20..37: BUFB = MK @ Wmt^T + bmt         [144,32]
        for (int job = wave; job < 38; job += 8) {
            v8f acc = {};
            if (job < 2) {
                int n0 = job * 16;
                wmma_tiles<4>(A2 + 128, 256, 0, Wax, 128, n0, lane, acc);
                store_d(afx, 32, 0, n0, lane, acc, Bax, 1);
            } else if (job < 20) {
                int jj = job - 2, m0 = (jj >> 1) * 16, n0 = (jj & 1) * 16;
                wmma_tiles<2>(TQ, 64, m0, Wat, 64, n0, lane, acc);
                store_d(BUFA, 32, m0, n0, lane, acc, Bat, 1);
            } else {
                int jj = job - 20, m0 = (jj >> 1) * 16, n0 = (jj & 1) * 16;
                wmma_tiles<6>(MK, 192, m0, Wmt, 192, n0, lane, acc);
                store_d(BUFB, 32, m0, n0, lane, acc, Bmt, 0);
            }
        }
        __syncthreads();

        // pt = sigmoid(sum(aftert * mt))
        if (tid < 144) {
            float s = 0.f;
#pragma unroll 8
            for (int o = 0; o < 32; ++o) s += BUFA[tid * 32 + o] * BUFB[tid * 32 + o];
            int r = tid / 9, kk = tid - r * 9;
            out[(size_t)(B_TOT * L_T * ADJ_K) +
                ((size_t)(bb + r) * L_T + t) * ADJ_K + kk] = 1.f / (1.f + __expf(-s));
        }
        __syncthreads();

        // BUFB <- MK @ Wmx^T + bmx
        for (int job = wave; job < 18; job += 8) {
            int m0 = (job >> 1) * 16, n0 = (job & 1) * 16;
            v8f acc = {};
            wmma_tiles<6>(MK, 192, m0, Wmx, 192, n0, lane, acc);
            store_d(BUFB, 32, m0, n0, lane, acc, Bmx, 0);
        }
        __syncthreads();

        // px = dot(afterx, mx) with adjacency mask
        if (tid < 144) {
            int r = tid / 9;
            float s = 0.f;
#pragma unroll 8
            for (int o = 0; o < 32; ++o) s += afx[r * 32 + o] * BUFB[tid * 32 + o];
            if (adjsh[tid] == N_EDGES) s = -__builtin_inff();
            pxs[tid] = s;
        }
        __syncthreads();

        // log_softmax over K=9 per batch row, write pred_x
        if (tid < 16) {
            float m = -__builtin_inff();
            for (int kk = 0; kk < 9; ++kk) m = fmaxf(m, pxs[tid * 9 + kk]);
            float se = 0.f;
            for (int kk = 0; kk < 9; ++kk) se += __expf(pxs[tid * 9 + kk] - m);
            float lse = m + __logf(se);
            size_t row = ((size_t)(bb + tid) * L_T + t) * ADJ_K;
            for (int kk = 0; kk < 9; ++kk) out[row + kk] = pxs[tid * 9 + kk] - lse;
        }
        __syncthreads();
    }
}

// ---------------- host launch ----------------
extern "C" void kernel_launch(void* const* d_in, const int* in_sizes, int n_in,
                              void* d_out, int out_size, void* d_ws, size_t ws_size,
                              hipStream_t stream) {
    const int*   x         = (const int*)d_in[0];
    const int*   dpt       = (const int*)d_in[1];
    const float* z         = (const float*)d_in[2];
    const int*   adj       = (const int*)d_in[3];
    const float* eprop     = (const float*)d_in[4];
    const float* x_emb     = (const float*)d_in[5];
    const float* t_emb     = (const float*)d_in[6];
    const float* tcost_emb = (const float*)d_in[7];
    const float* w_ih0 = (const float*)d_in[8],  *w_hh0 = (const float*)d_in[9];
    const float* b_ih0 = (const float*)d_in[10], *b_hh0 = (const float*)d_in[11];
    const float* w_ih1 = (const float*)d_in[12], *w_hh1 = (const float*)d_in[13];
    const float* b_ih1 = (const float*)d_in[14], *b_hh1 = (const float*)d_in[15];
    const float* w_ih2 = (const float*)d_in[16], *w_hh2 = (const float*)d_in[17];
    const float* b_ih2 = (const float*)d_in[18], *b_hh2 = (const float*)d_in[19];
    const float* wax = (const float*)d_in[20], *bax = (const float*)d_in[21];
    const float* wat = (const float*)d_in[22], *bat = (const float*)d_in[23];
    const float* wmx = (const float*)d_in[24], *bmx = (const float*)d_in[25];
    const float* wmt = (const float*)d_in[26], *bmt = (const float*)d_in[27];

    prep_kernel<<<512, 256, 0, stream>>>(
        w_ih0, w_hh0, b_ih0, b_hh0, w_ih1, w_hh1, b_ih1, b_hh1,
        w_ih2, w_hh2, b_ih2, b_hh2, wax, bax, wat, bat, wmx, bmx, wmt, bmt,
        (char*)d_ws);

    model_kernel<<<B_TOT / 16, 256, SMEM_BYTES, stream>>>(
        x, dpt, z, adj, eprop, x_emb, t_emb, tcost_emb,
        (const char*)d_ws, (float*)d_out);
}
